// QuantizedConv2d_9852654977323
// MI455X (gfx1250) — matrix operand order repaired and checked by reading, not compiled, not run on hardware
//
#include <hip/hip_runtime.h>

// ---------------- problem constants ----------------
#define O_CH   256
#define I_CH   128
#define KH_    3
#define KW_    3
#define HW_    112
#define NB_    16
#define KG     (I_CH * KH_ * KW_)          // 1152 (reordered as (kh,kw,c))
#define PLANE  (HW_ * HW_)                 // 12544
#define MT     (NB_ * PLANE)               // 200704 pixels (== 1568 * 128)

// ---------------- GEMM tiling ----------------
#define BM   128
#define BN   256                           // full O: A-operand read exactly once
#define BK   32
#define LDA  40   // halfs per LDS row (80B rows: 16B aligned, conflict-light)
#define LDB  40
#define NIT  (KG / BK)                     // 36 K-steps

typedef __attribute__((ext_vector_type(16))) _Float16 v16h;
typedef __attribute__((ext_vector_type(8)))  _Float16 v8h;
typedef __attribute__((ext_vector_type(4)))  _Float16 v4h;
typedef __attribute__((ext_vector_type(8)))  float    v8f;
typedef __attribute__((ext_vector_type(4)))  int      v4i;

#define GLOBAL_AS __attribute__((address_space(1)))
#define LDS_AS    __attribute__((address_space(3)))

#if __has_builtin(__builtin_amdgcn_global_load_async_to_lds_b128) && \
    __has_builtin(__builtin_amdgcn_s_wait_asynccnt)
#define HAVE_ASYNC 1
#else
#define HAVE_ASYNC 0
#endif

// =====================================================================
// Pass 0: x (f32 NCHW) -> f16, same layout. 4-wide.
// =====================================================================
__global__ __launch_bounds__(256) void cvt_x_kernel(
    const float* __restrict__ x, _Float16* __restrict__ x16)
{
    int i = blockIdx.x * blockDim.x + threadIdx.x;   // element-quad index
    const float4 f = ((const float4*)x)[i];
    v4h h = { (_Float16)f.x, (_Float16)f.y, (_Float16)f.z, (_Float16)f.w };
    ((v4h*)x16)[i] = h;
}

// =====================================================================
// Pass 1: unpack + dequantize weights -> f16 workspace, layout [O][KG]
// with k = (kh*3 + kw)*I_CH + c  (tap-major, channel-contiguous)
// =====================================================================
__global__ __launch_bounds__(256) void dequant_kernel(
    const float* __restrict__ packed, const float* __restrict__ scale,
    const float* __restrict__ zp, _Float16* __restrict__ Bws)
{
    int i = blockIdx.x * blockDim.x + threadIdx.x;
    const int NP = O_CH * I_CH * KH_ * KW_ / 2;      // 147456
    if (i >= NP) return;
    float p  = packed[i];
    float q1 = floorf(p);
    float q2 = fminf(fmaxf(rintf((p - q1) * 255.0f), 0.0f), 255.0f);
    float s = scale[0], z = zp[0];
#pragma unroll
    for (int t = 0; t < 2; ++t) {
        int   f = 2 * i + t;
        float q = t ? q2 : q1;
        int kw  = f % KW_;  int tmp = f / KW_;
        int kh  = tmp % KH_; tmp /= KH_;
        int c   = tmp % I_CH;
        int o   = tmp / I_CH;
        Bws[(size_t)o * KG + (kh * KW_ + kw) * I_CH + c] = (_Float16)(q * s + z);
    }
}

// =====================================================================
// Pass 2: implicit-GEMM conv via v_wmma_f32_16x16x32_f16
//   M = pixels, N = 256 channels, K = 1152; block 128x256x32,
//   8 waves (4M x 2N), wave tile 32x128 -> 16 WMMAs / K-step.
//   Double-buffered LDS; B tile staged with async-to-LDS engine.
// =====================================================================
template <typename TX>
__global__ __launch_bounds__(256) void conv_wmma_kernel(
    const TX*       __restrict__ x,     // [NB][I_CH][HW][HW] (f32 or f16)
    const _Float16* __restrict__ Bws,   // [O_CH][KG] f16
    const float*    __restrict__ bias,  // [O_CH]
    float*          __restrict__ out)   // [NB][O_CH][HW][HW] f32
{
    __shared__ union SMem {
        struct { _Float16 As[2][BM][LDA]; _Float16 Bs[2][BN][LDB]; } t;
        float cs[8][16][33];
    } smem;

    const int t     = threadIdx.x;
    const int wave  = t >> 5;
    const int lane  = t & 31;
    const int wm    = wave & 3;          // M sub-tile (0..3)
    const int wn    = wave >> 2;         // N sub-tile (0..1)
    const int l15   = lane & 15;
    const int halfl = lane >> 4;

    const int m0 = blockIdx.x * BM;      // 1568 tiles, exact

    // ---- A loader: thread owns row mA, k-halves kA..kA+15
    const int mA  = t & 127;
    const int kA  = (t >> 7) * 16;
    const int mAg = m0 + mA;
    const int nimgA = mAg / PLANE;
    const int remA  = mAg % PLANE;
    const int ohA   = remA / HW_;
    const int owA   = remA % HW_;

    // ---- B loader: thread t owns channel row t (64B contiguous per chunk)
    const _Float16* bsrc = Bws + (size_t)t * KG;

    // ---------- staging helpers ----------
    auto stageA = [&](int buf, int kk) {
        const int p   = kk >> 7;         // filter tap 0..8, fixed per chunk
        const int c0  = kk & 127;
        const int kh  = p / 3;
        const int kwp = p % 3;
        const int ih  = ohA + kh - 1;
        const int iw  = owA + kwp - 1;
        const bool valid = ((unsigned)ih < HW_) && ((unsigned)iw < HW_);
        const TX* xp =
            x + (((size_t)(nimgA * I_CH + c0 + kA) * HW_ + ih) * HW_ + iw);
#pragma unroll
        for (int j = 0; j < 16; ++j)
            smem.t.As[buf][mA][kA + j] = valid ? (_Float16)xp[(size_t)j * PLANE]
                                               : (_Float16)0.0f;
        if (valid) __builtin_prefetch(xp + (size_t)16 * PLANE, 0, 0);
    };
    auto stageB = [&](int buf, int kk) {
#if HAVE_ASYNC
        GLOBAL_AS v4i* g = (GLOBAL_AS v4i*)(bsrc + kk);
        LDS_AS    v4i* l = (LDS_AS v4i*)(&smem.t.Bs[buf][t][0]);
#pragma unroll
        for (int seg = 0; seg < 4; ++seg)
            __builtin_amdgcn_global_load_async_to_lds_b128(g + seg, l + seg, 0, 0);
#else
        const v8h* s = (const v8h*)(bsrc + kk);
        v8h b0 = s[0], b1 = s[1], b2 = s[2], b3 = s[3];
        *(v8h*)(&smem.t.Bs[buf][t][0])  = b0;
        *(v8h*)(&smem.t.Bs[buf][t][8])  = b1;
        *(v8h*)(&smem.t.Bs[buf][t][16]) = b2;
        *(v8h*)(&smem.t.Bs[buf][t][24]) = b3;
#endif
    };

    union AF { v16h v; unsigned u[8]; };
    v8f acc[2][8];
#pragma unroll
    for (int fm = 0; fm < 2; ++fm)
#pragma unroll
        for (int fn = 0; fn < 8; ++fn)
            acc[fm][fn] = (v8f){};

    // ---------- pipelined main loop (1 barrier / K-step) ----------
    stageA(0, 0);
    stageB(0, 0);
#if HAVE_ASYNC
    __builtin_amdgcn_s_wait_asynccnt(0);
#endif
    __syncthreads();

    for (int it = 0; it < NIT; ++it) {
        const int cur = it & 1;
        if (it + 1 < NIT) {              // stage next tile into alternate buffer
            stageA(cur ^ 1, (it + 1) * BK);
            stageB(cur ^ 1, (it + 1) * BK);
        }

        // fragments + 16 WMMAs from current buffer
        AF af[2];
#pragma unroll
        for (int fm = 0; fm < 2; ++fm) {
            const unsigned* arow =
                (const unsigned*)(&smem.t.As[cur][wm * 32 + fm * 16 + l15][0]);
#pragma unroll
            for (int v = 0; v < 8; ++v)
                af[fm].u[v] = arow[v + (v & 4) + halfl * 4];
        }
#pragma unroll
        for (int fn = 0; fn < 8; ++fn) {
            AF bf;
            const unsigned* brow =
                (const unsigned*)(&smem.t.Bs[cur][wn * 128 + fn * 16 + l15][0]);
#pragma unroll
            for (int v = 0; v < 8; ++v)
                bf.u[v] = brow[v + (v & 4) + halfl * 4];
            acc[0][fn] = __builtin_amdgcn_wmma_f32_16x16x32_f16(
                false, af[0].v, false, bf.v, (short)0, acc[0][fn], false, false);
            acc[1][fn] = __builtin_amdgcn_wmma_f32_16x16x32_f16(
                false, af[1].v, false, bf.v, (short)0, acc[1][fn], false, false);
        }

#if HAVE_ASYNC
        __builtin_amdgcn_s_wait_asynccnt(0);   // next tile's async copies done
#endif
        __syncthreads();                       // everyone done reading buf cur
    }

    // ---------- epilogue: LDS transpose -> coalesced 128B stores ----------
    const int mS    = m0 + wm * 32 + lane;
    const int nimgS = mS / PLANE;
    const int remS  = mS % PLANE;
    const size_t outBase =
        (size_t)nimgS * O_CH * PLANE + (size_t)(remS / HW_) * HW_ + (remS % HW_);
    float (*cbuf)[33] = smem.cs[wave];

#pragma unroll
    for (int fn = 0; fn < 8; ++fn) {
#pragma unroll
        for (int fm = 0; fm < 2; ++fm)
#pragma unroll
            for (int r = 0; r < 8; ++r)
                cbuf[l15][fm * 16 + halfl * 8 + r] = acc[fm][fn][r];
        __syncthreads();
#pragma unroll
        for (int nn = 0; nn < 16; ++nn) {
            const int oN = wn * 128 + fn * 16 + nn;
            out[outBase + (size_t)oN * PLANE] = cbuf[nn][lane] + bias[oN];
        }
        __syncthreads();
    }
}

// =====================================================================
extern "C" void kernel_launch(void* const* d_in, const int* in_sizes, int n_in,
                              void* d_out, int out_size, void* d_ws, size_t ws_size,
                              hipStream_t stream) {
    const float* x      = (const float*)d_in[0];
    const float* packed = (const float*)d_in[1];
    const float* scale  = (const float*)d_in[2];
    const float* zp     = (const float*)d_in[3];
    const float* bias   = (const float*)d_in[4];
    float*       outp   = (float*)d_out;

    const size_t wbytes = (size_t)O_CH * KG * sizeof(_Float16);   // 576 KB
    const size_t xelems = (size_t)NB_ * I_CH * PLANE;             // 25.7M
    const size_t xbytes = xelems * sizeof(_Float16);              // ~49 MB

    _Float16* Bws = (_Float16*)d_ws;
    const int NP = O_CH * I_CH * KH_ * KW_ / 2;
    dequant_kernel<<<(NP + 255) / 256, 256, 0, stream>>>(packed, scale, zp, Bws);

    dim3 grid(MT / BM);   // 1568

    if (ws_size >= wbytes + xbytes) {
        _Float16* x16 = (_Float16*)((char*)d_ws + wbytes);        // 256B-mult offset
        cvt_x_kernel<<<(int)(xelems / 4 / 256), 256, 0, stream>>>(x, x16);
        conv_wmma_kernel<_Float16><<<grid, 256, 0, stream>>>(x16, Bws, bias, outp);
    } else {
        conv_wmma_kernel<float><<<grid, 256, 0, stream>>>(x, Bws, bias, outp);
    }
}
